// CausalAttention_39556648796514
// MI455X (gfx1250) — compile-verified, matrix operand order checked by reading, not run on hardware
//
#include <hip/hip_runtime.h>

// CDNA5 / gfx1250 WMMA + vector types
typedef __attribute__((ext_vector_type(16))) _Float16 v16h;
typedef __attribute__((ext_vector_type(8)))  _Float16 v8h;
typedef __attribute__((ext_vector_type(8)))  float    v8f;
typedef __attribute__((ext_vector_type(4)))  float    v4f;

#define T_  2048
#define D_  1024
#define H_  16
#define DH_ 64

// ---------------------------------------------------------------------------
// f16 fragment loader (A and B for C = X @ W^T share this addressing).
// ISA 7.12.2 16-bit 16x32 A layout: lane L -> row M=L&15;
//   lanes 0-15 : K = k0+0..7 and k0+16..23
//   lanes 16-31: K = k0+8..15 and k0+24..31
// Two guaranteed-aligned 16B loads -> global_load_b128 / ds_load_b128.
// ---------------------------------------------------------------------------
static __device__ __forceinline__ v16h load_frag_f16(const _Float16* src,
                                                     int ld, int row0, int k0) {
  const int lane = threadIdx.x & 31;
  const int r    = row0 + (lane & 15);
  const int kb   = k0 + ((lane >> 4) << 3);
  const _Float16* p = src + (size_t)r * ld + kb;
  v8h lo = *(const v8h*)p;
  v8h hi = *(const v8h*)(p + 16);
  return __builtin_shufflevector(lo, hi, 0, 1, 2, 3, 4, 5, 6, 7,
                                 8, 9, 10, 11, 12, 13, 14, 15);
}

static __device__ __forceinline__ v8f zero8() {
  v8f z;
#pragma unroll
  for (int i = 0; i < 8; ++i) z[i] = 0.0f;
  return z;
}

static __device__ __forceinline__ v8f wmma_f16(v16h a, v16h b, v8f c) {
  return __builtin_amdgcn_wmma_f32_16x16x32_f16(false, a, false, b,
                                                (short)0, c, false, false);
}

// ---------------------------------------------------------------------------
// Kernel 0: one-shot f32 -> f16 conversion (x, w_kqv, w_proj). 8 elems/thread.
// ---------------------------------------------------------------------------
__global__ __launch_bounds__(256) void CausalAttention_cvt_kernel(
    const float* __restrict__ src, _Float16* __restrict__ dst, int n) {
  const int base = (int)(blockIdx.x * blockDim.x + threadIdx.x) * 8;
  if (base >= n) return;
  v4f a = *(const v4f*)(src + base);
  v4f b = *(const v4f*)(src + base + 4);
  v8h h;
#pragma unroll
  for (int i = 0; i < 4; ++i) h[i] = (_Float16)a[i];
#pragma unroll
  for (int i = 0; i < 4; ++i) h[4 + i] = (_Float16)b[i];
  *(v8h*)(dst + base) = h;
}

// ---------------------------------------------------------------------------
// Kernel 1: kqv = x @ w_kqv^T + b_kqv, scattered into per-head f16 buffers:
//   K[h][t][dh], Q[h][t][dh] (x sqrt(dh)=8 folded in), Vt[h][dh][t].
// Column mapping (reference): c -> which=c/D, d=c%D, h=d%H, dh_idx=d/H.
// One wave: 32(M) x 64(N) tile = 8 WMMAs per k-step vs 6 fragment loads.
// ---------------------------------------------------------------------------
__global__ __launch_bounds__(256) void CausalAttention_kqv_kernel(
    const _Float16* __restrict__ x16, const _Float16* __restrict__ w16,
    const float* __restrict__ bkqv,
    _Float16* __restrict__ Kb, _Float16* __restrict__ Qb,
    _Float16* __restrict__ Vt) {
  const int wave = (int)((blockIdx.x * blockDim.x + threadIdx.x) >> 5);
  const int NSTRIPS = (3 * D_) / 64;  // 48
  if (wave >= (T_ / 32) * NSTRIPS) return;
  const int m0   = (wave / NSTRIPS) * 32;
  const int col0 = (wave % NSTRIPS) * 64;
  const int lane = threadIdx.x & 31;
  const int hi   = lane >> 4;
  const int n    = lane & 15;

  v8f acc[2][4];
#pragma unroll
  for (int mi = 0; mi < 2; ++mi)
#pragma unroll
    for (int j = 0; j < 4; ++j) acc[mi][j] = zero8();

  for (int k0 = 0; k0 < D_; k0 += 32) {
    v16h a0 = load_frag_f16(x16, D_, m0, k0);
    v16h a1 = load_frag_f16(x16, D_, m0 + 16, k0);
    if (k0 + 32 < D_) {  // global_prefetch_b8 of next A k-tile
      __builtin_prefetch(x16 + (size_t)(m0 + n) * D_ + k0 + 32, 0, 3);
    }
#pragma unroll
    for (int j = 0; j < 4; ++j) {
      v16h b = load_frag_f16(w16, D_, col0 + j * 16, k0);
      acc[0][j] = wmma_f16(a0, b, acc[0][j]);
      acc[1][j] = wmma_f16(a1, b, acc[1][j]);
    }
  }

#pragma unroll
  for (int mi = 0; mi < 2; ++mi) {
    const int tbase = m0 + mi * 16 + hi * 8;  // lanes>=16 hold rows M=8..15
#pragma unroll
    for (int j = 0; j < 4; ++j) {
      const int c     = col0 + j * 16 + n;  // [0, 3072)
      const int which = c >> 10;            // 0=K, 1=Q, 2=V (reference order)
      const int d     = c & (D_ - 1);
      const int h     = d & (H_ - 1);
      const int dd    = d >> 4;
      const float bias = bkqv[c];
      if (which == 2) {  // transposed V: 8 consecutive t per lane -> one b128
        v8h pk;
#pragma unroll
        for (int i = 0; i < 8; ++i) pk[i] = (_Float16)(acc[mi][j][i] + bias);
        *(v8h*)(Vt + (size_t)(h * DH_ + dd) * T_ + tbase) = pk;
      } else {
        _Float16* dstb = (which == 0) ? Kb : Qb;
        const float s  = (which == 0) ? 1.0f : 8.0f;  // fold sqrt(dh) into Q
#pragma unroll
        for (int i = 0; i < 8; ++i)
          dstb[((size_t)h * T_ + tbase + i) * DH_ + dd] =
              (_Float16)((acc[mi][j][i] + bias) * s);
      }
    }
  }
}

// ---------------------------------------------------------------------------
// Kernel 2: flash-style causal attention. One wave owns a 16-query tile of
// one head; streams 32-key tiles with online softmax; O accumulated in f32.
// ---------------------------------------------------------------------------
__global__ __launch_bounds__(256) void CausalAttention_attn_kernel(
    const _Float16* __restrict__ Kb, const _Float16* __restrict__ Qb,
    const _Float16* __restrict__ Vt, _Float16* __restrict__ Mg) {
  __shared__ _Float16 lds[8][16 * 40];  // per-wave 16x32 P tile, stride 40 halfs
  const int wslot = threadIdx.x >> 5;
  const int wave  = (int)(blockIdx.x * 8 + wslot);
  if (wave >= H_ * (T_ / 16)) return;
  const int h    = wave >> 7;  // 128 q-tiles per head
  const int q0   = (wave & 127) * 16;
  const int lane = threadIdx.x & 31;
  const int hi   = lane >> 4;
  const int n    = lane & 15;

  const _Float16* Kh = Kb + (size_t)h * T_ * DH_;
  const _Float16* Qh = Qb + (size_t)h * T_ * DH_;
  const _Float16* Vh = Vt + (size_t)h * DH_ * T_;

  // Q A-fragments are loop invariant (dh = 64 -> two K=32 fragments).
  const v16h qa0  = load_frag_f16(Qh, DH_, q0, 0);
  const v16h qa32 = load_frag_f16(Qh, DH_, q0, 32);

  v8f o[4];
#pragma unroll
  for (int j = 0; j < 4; ++j) o[j] = zero8();
  float m[8], l[8];
#pragma unroll
  for (int i = 0; i < 8; ++i) { m[i] = -1e30f; l[i] = 0.0f; }

  _Float16* P = &lds[wslot][0];
  const int nkt = (q0 + 16 + 31) >> 5;  // causal bound: keys 0..q0+15

  for (int kt = 0; kt < nkt; ++kt) {
    const int kt0 = kt * 32;
    if (kt + 1 < nkt) {  // prefetch next K tile rows
      __builtin_prefetch(Kh + (size_t)(kt0 + 32 + lane) * DH_, 0, 3);
    }
    // S = Q @ K^T for 32 keys: two 16x16 score frags, dh reduced in 2 steps
    v8f s0 = zero8(), s1 = zero8();
    {
      v16h b;
      b = load_frag_f16(Kh, DH_, kt0, 0);       s0 = wmma_f16(qa0,  b, s0);
      b = load_frag_f16(Kh, DH_, kt0, 32);      s0 = wmma_f16(qa32, b, s0);
      b = load_frag_f16(Kh, DH_, kt0 + 16, 0);  s1 = wmma_f16(qa0,  b, s1);
      b = load_frag_f16(Kh, DH_, kt0 + 16, 32); s1 = wmma_f16(qa32, b, s1);
    }
    if (kt0 + 31 > q0) {  // only the diagonal tile needs masking
#pragma unroll
      for (int i = 0; i < 8; ++i) {
        const int r = q0 + hi * 8 + i;
        if (kt0 + n      > r) s0[i] = -1e30f;
        if (kt0 + 16 + n > r) s1[i] = -1e30f;
      }
    }
    // Online softmax: row stats live in the 16 lanes of each half-wave.
#pragma unroll
    for (int i = 0; i < 8; ++i) {
      float rmax = fmaxf(s0[i], s1[i]);
      rmax = fmaxf(rmax, __shfl_xor(rmax, 1, 32));
      rmax = fmaxf(rmax, __shfl_xor(rmax, 2, 32));
      rmax = fmaxf(rmax, __shfl_xor(rmax, 4, 32));
      rmax = fmaxf(rmax, __shfl_xor(rmax, 8, 32));
      const float mnew  = fmaxf(m[i], rmax);
      const float alpha = __expf(m[i] - mnew);
      const float p0 = __expf(s0[i] - mnew);
      const float p1 = __expf(s1[i] - mnew);
      float rsum = p0 + p1;
      rsum += __shfl_xor(rsum, 1, 32);
      rsum += __shfl_xor(rsum, 2, 32);
      rsum += __shfl_xor(rsum, 4, 32);
      rsum += __shfl_xor(rsum, 8, 32);
      l[i] = l[i] * alpha + rsum;
      m[i] = mnew;
#pragma unroll
      for (int j = 0; j < 4; ++j) o[j][i] *= alpha;
      // D-layout -> row-major LDS (row = hi*8+i, cols n and 16+n)
      P[(hi * 8 + i) * 40 + n]      = (_Float16)p0;
      P[(hi * 8 + i) * 40 + 16 + n] = (_Float16)p1;
    }
    // Re-load P as a 16x32 A-fragment; O += P @ V (B from transposed V).
    const v16h pa = load_frag_f16(P, 40, 0, 0);
#pragma unroll
    for (int j = 0; j < 4; ++j) {
      v16h b = load_frag_f16(Vh, T_, j * 16, kt0);
      o[j] = wmma_f16(pa, b, o[j]);
    }
  }

  // merged[t][h*64 + dcol], f16 for the projection GEMM
#pragma unroll
  for (int i = 0; i < 8; ++i) {
    const float inv = 1.0f / l[i];
    const int t = q0 + hi * 8 + i;
#pragma unroll
    for (int j = 0; j < 4; ++j) {
      Mg[(size_t)t * D_ + h * DH_ + j * 16 + n] = (_Float16)(o[j][i] * inv);
    }
  }
}

// ---------------------------------------------------------------------------
// Kernel 3: out = merged @ w_proj^T + b_proj (f32 output). 32x64 per wave.
// ---------------------------------------------------------------------------
__global__ __launch_bounds__(256) void CausalAttention_proj_kernel(
    const _Float16* __restrict__ Mg, const _Float16* __restrict__ wp16,
    const float* __restrict__ bp, float* __restrict__ out) {
  const int wave = (int)((blockIdx.x * blockDim.x + threadIdx.x) >> 5);
  const int NSTRIPS = D_ / 64;  // 16
  if (wave >= (T_ / 32) * NSTRIPS) return;
  const int m0   = (wave / NSTRIPS) * 32;
  const int col0 = (wave % NSTRIPS) * 64;
  const int lane = threadIdx.x & 31;
  const int hi   = lane >> 4;
  const int n    = lane & 15;

  v8f acc[2][4];
#pragma unroll
  for (int mi = 0; mi < 2; ++mi)
#pragma unroll
    for (int j = 0; j < 4; ++j) acc[mi][j] = zero8();

  for (int k0 = 0; k0 < D_; k0 += 32) {
    v16h a0 = load_frag_f16(Mg, D_, m0, k0);
    v16h a1 = load_frag_f16(Mg, D_, m0 + 16, k0);
    if (k0 + 32 < D_) {
      __builtin_prefetch(Mg + (size_t)(m0 + n) * D_ + k0 + 32, 0, 3);
    }
#pragma unroll
    for (int j = 0; j < 4; ++j) {
      v16h b = load_frag_f16(wp16, D_, col0 + j * 16, k0);
      acc[0][j] = wmma_f16(a0, b, acc[0][j]);
      acc[1][j] = wmma_f16(a1, b, acc[1][j]);
    }
  }

#pragma unroll
  for (int mi = 0; mi < 2; ++mi) {
#pragma unroll
    for (int j = 0; j < 4; ++j) {
      const int c = col0 + j * 16 + n;
      const float bias = bp[c];
#pragma unroll
      for (int i = 0; i < 8; ++i) {
        const int t = m0 + mi * 16 + hi * 8 + i;
        out[(size_t)t * D_ + c] = acc[mi][j][i] + bias;
      }
    }
  }
}

// ---------------------------------------------------------------------------
extern "C" void kernel_launch(void* const* d_in, const int* in_sizes, int n_in,
                              void* d_out, int out_size, void* d_ws, size_t ws_size,
                              hipStream_t stream) {
  const float* x     = (const float*)d_in[0];
  const float* wkqv  = (const float*)d_in[1];
  const float* bkqv  = (const float*)d_in[2];
  const float* wproj = (const float*)d_in[3];
  const float* bproj = (const float*)d_in[4];
  // d_in[5] = n_heads == 16, baked in as H_.

  // Workspace layout (f16 halves):
  //   x16   : T*D          = 2,097,152
  //   w16   : 3D*D         = 3,145,728
  //   wp16  : D*D          = 1,048,576
  //   K,Q,Vt: H*T*DH each  = 2,097,152 each
  //   Mg    : T*D          = 2,097,152   -> total 28 MB
  _Float16* x16  = (_Float16*)d_ws;
  _Float16* w16  = x16 + (size_t)T_ * D_;
  _Float16* wp16 = w16 + (size_t)3 * D_ * D_;
  _Float16* Kb   = wp16 + (size_t)D_ * D_;
  _Float16* Qb   = Kb + (size_t)H_ * T_ * DH_;
  _Float16* Vt   = Qb + (size_t)H_ * T_ * DH_;
  _Float16* Mg   = Vt + (size_t)H_ * T_ * DH_;
  float* out = (float*)d_out;

  // Kernel 0: one-shot f32->f16 conversions (8 elems/thread).
  CausalAttention_cvt_kernel<<<(T_ * D_) / (8 * 256), 256, 0, stream>>>(
      x, x16, T_ * D_);
  CausalAttention_cvt_kernel<<<(3 * D_ * D_) / (8 * 256), 256, 0, stream>>>(
      wkqv, w16, 3 * D_ * D_);
  CausalAttention_cvt_kernel<<<(D_ * D_) / (8 * 256), 256, 0, stream>>>(
      wproj, wp16, D_ * D_);

  // Kernel 1: 64 M-strips x 48 N-strips = 3072 waves -> 384 blocks x 8 waves.
  CausalAttention_kqv_kernel<<<384, 256, 0, stream>>>(x16, w16, bkqv,
                                                      Kb, Qb, Vt);
  // Kernel 2: 16 heads x 128 q-tiles = 2048 waves -> 256 blocks.
  CausalAttention_attn_kernel<<<256, 256, 0, stream>>>(Kb, Qb, Vt, Mg);
  // Kernel 3: 64 M-strips x 16 N-strips = 1024 waves -> 128 blocks.
  CausalAttention_proj_kernel<<<128, 256, 0, stream>>>(Mg, wp16, bproj, out);
}
